// SelfAttention_44444321579671
// MI455X (gfx1250) — matrix operand order repaired and checked by reading, not compiled, not run on hardware
//
#include <hip/hip_runtime.h>
#include <math.h>

// ---------------------------------------------------------------------------
// Types for CDNA5 WMMA (wave32): v_wmma_f32_16x16x32_bf16
// ---------------------------------------------------------------------------
typedef __attribute__((ext_vector_type(16))) __bf16 v16bf;
typedef __attribute__((ext_vector_type(2)))  __bf16 bf2;
typedef __attribute__((ext_vector_type(8)))  float  v8f;

constexpr int kNI = 512;     // channels
constexpr int kS  = 1024;    // sequence length
constexpr int kN  = 16;      // batch
constexpr int kH  = 8;       // heads
constexpr int kD  = 64;      // head dim
constexpr float kScaleInv = 0.125f;   // 1/sqrt(64)
constexpr float kLnEps    = 1e-5f;

// ---------------------------------------------------------------------------
// Reductions (wave32; rows of a WMMA D tile live in a 16-lane half)
// ---------------------------------------------------------------------------
__device__ inline float wave_reduce_add(float v) {
#pragma unroll
  for (int m = 1; m < 32; m <<= 1) v += __shfl_xor(v, m, 32);
  return v;
}
__device__ inline float half_reduce_add(float v) {
#pragma unroll
  for (int m = 1; m < 16; m <<= 1) v += __shfl_xor(v, m, 32);
  return v;
}
__device__ inline float half_reduce_max(float v) {
#pragma unroll
  for (int m = 1; m < 16; m <<= 1) v = fmaxf(v, __shfl_xor(v, m, 32));
  return v;
}

// ---------------------------------------------------------------------------
// Fragment loaders (this code's contiguous k order, used for both A and B)
// A (16x32 bf16, MxK, row-major src): lane=M%16, kbase = lane<16 ? 0 : 16,
//   VGPR v holds K = kbase+2v,2v+1.
// B (32x16 bf16, KxN) loaded from *transposed* storage Bt[N][K]:
//   lane=N%16, kbase = lane<16 ? 0 : 16, element e holds K = kbase+e.
// D (16x16 f32): lane holds col N=lane%16; VGPR r holds row M=r+8*(lane/16).
// ---------------------------------------------------------------------------
__device__ inline v16bf load_a_frag(const __bf16* A, int lda) {
  const int lane = threadIdx.x & 31;
  const int kb   = (lane < 16) ? 0 : 16;
  const __bf16* row = A + (lane & 15) * lda;
  v16bf a;
#pragma unroll
  for (int v = 0; v < 8; ++v) {
    const int k = kb + 2 * v;
    bf2 t = *(const bf2*)(row + k);
    a[2 * v]     = t.x;
    a[2 * v + 1] = t.y;
  }
  return a;
}

__device__ inline v16bf load_bt_frag(const __bf16* Bt, int ldb) {
  const int lane = threadIdx.x & 31;
  const __bf16* p = Bt + (lane & 15) * ldb + ((lane < 16) ? 0 : 16);
  return *(const v16bf*)p;   // 32B contiguous -> global_load_b128 pair
}

__device__ inline v8f wmma_bf16(v16bf a, v16bf b, v8f c) {
  return __builtin_amdgcn_wmma_f32_16x16x32_bf16(
      /*neg_a=*/false, a, /*neg_b=*/false, b,
      /*c_mod=*/(short)0, c, /*reuse_a=*/false, /*reuse_b=*/false);
}

// ---------------------------------------------------------------------------
// Kernel 1: fused transpose + LayerNorm + bf16 cast.
// x[n][c][s] -> xn[(n*s+srow)][c] (bf16). One wave per row; x fits in L2.
// ---------------------------------------------------------------------------
__global__ void ln_kernel(const float* __restrict__ x,
                          const float* __restrict__ g,
                          const float* __restrict__ b,
                          __bf16* __restrict__ xn) {
  const int wave = threadIdx.x >> 5;
  const int lane = threadIdx.x & 31;
  const int row  = blockIdx.x * (blockDim.x >> 5) + wave;   // n*1024 + srow
  if (row >= kN * kS) return;
  const int n = row >> 10, srow = row & (kS - 1);
  const float* xb = x + (size_t)n * kNI * kS + srow;        // stride kS per c
  float v[16];
  float s1 = 0.f, s2 = 0.f;
#pragma unroll
  for (int j = 0; j < 16; ++j) {
    const float t = xb[(size_t)(lane + 32 * j) * kS];
    v[j] = t; s1 += t; s2 += t * t;
  }
  s1 = wave_reduce_add(s1);
  s2 = wave_reduce_add(s2);
  const float mu   = s1 * (1.f / kNI);
  const float var  = s2 * (1.f / kNI) - mu * mu;
  const float rstd = rsqrtf(var + kLnEps);
  __bf16* o = xn + (size_t)row * kNI;
#pragma unroll
  for (int j = 0; j < 16; ++j) {
    const int c = lane + 32 * j;
    o[c] = (__bf16)((v[j] - mu) * rstd * g[c] + b[c]);
  }
}

// ---------------------------------------------------------------------------
// Kernel 2: weight convert + transpose to bf16 Bt[N][K] layout.
// ---------------------------------------------------------------------------
__global__ void wconv_kernel(const float* __restrict__ qkv_w,
                             const float* __restrict__ proj_w,
                             __bf16* __restrict__ qkv_wt,
                             __bf16* __restrict__ proj_wt) {
  const int idx = blockIdx.x * blockDim.x + threadIdx.x;
  const int NQ = kNI * 3 * kNI;
  if (idx < NQ) {
    const int i = idx / (3 * kNI), c = idx % (3 * kNI);
    qkv_wt[(size_t)c * kNI + i] = (__bf16)qkv_w[idx];
  } else {
    const int j = idx - NQ;
    if (j < kNI * kNI) {
      const int i = j / kNI, c = j % kNI;
      proj_wt[(size_t)c * kNI + i] = (__bf16)proj_w[j];
    }
  }
}

// ---------------------------------------------------------------------------
// Scatter one 16x16 f32 D tile of the QKV GEMM into Q/K/V^T (bf16) + bias.
// ---------------------------------------------------------------------------
__device__ inline void qkv_store_tile(const v8f& acc, int mbase, int c,
                                      const float* __restrict__ qkv_b,
                                      __bf16* __restrict__ Qb,
                                      __bf16* __restrict__ Kb,
                                      __bf16* __restrict__ Vt) {
  const int lane = threadIdx.x & 31;
  const int h = c / 192, w = c % 192;
  const float bias = qkv_b[c];
#pragma unroll
  for (int r = 0; r < 8; ++r) {
    const int mg   = mbase + r + ((lane >> 4) << 3);
    const int n    = mg >> 10, srow = mg & (kS - 1);
    const __bf16 val = (__bf16)(acc[r] + bias);
    const size_t nh = (size_t)(n * kH + h);
    if (w < 64)        Qb[(nh * kS + srow) * kD + w]          = val;
    else if (w < 128)  Kb[(nh * kS + srow) * kD + (w - 64)]   = val;
    else               Vt[(nh * kD + (w - 128)) * kS + srow]  = val;
  }
}

// ---------------------------------------------------------------------------
// Kernel 3: QKV GEMM (16384x512 @ 512x1536) on bf16 WMMA.
// 2x2 register blocking: one 32x32 D block per wave (4 WMMAs share 2 A + 2 B
// fragment loads per K-step -> 2 b128-pairs per WMMA instead of 4).
// Fused bias + scatter into Q[n,h,s,d], K[n,h,s,d], V^T[n,h,d,s].
// ---------------------------------------------------------------------------
__global__ void __launch_bounds__(128)
qkv_gemm_kernel(const __bf16* __restrict__ xn, const __bf16* __restrict__ wt,
                const float* __restrict__ qkv_b,
                __bf16* __restrict__ Qb, __bf16* __restrict__ Kb,
                __bf16* __restrict__ Vt) {
  const int wid = blockIdx.x * (blockDim.x >> 5) + (threadIdx.x >> 5);
  const int NTC = (3 * kNI) / 32;                 // 48 col blocks
  const int rt = wid / NTC, ct = wid % NTC;
  if (rt >= (kN * kS) / 32) return;
  const int lane = threadIdx.x & 31;
  const __bf16* arow0 = xn + (size_t)(rt * 32) * kNI;
  const __bf16* arow1 = arow0 + (size_t)16 * kNI;
  const __bf16* brow0 = wt + (size_t)(ct * 32) * kNI;
  const __bf16* brow1 = brow0 + (size_t)16 * kNI;
  v8f acc00 = {}, acc01 = {}, acc10 = {}, acc11 = {};
#pragma unroll 2
  for (int k0 = 0; k0 < kNI; k0 += 32) {
    const v16bf a0 = load_a_frag(arow0 + k0, kNI);
    const v16bf a1 = load_a_frag(arow1 + k0, kNI);
    const v16bf b0 = load_bt_frag(brow0 + k0, kNI);
    const v16bf b1 = load_bt_frag(brow1 + k0, kNI);
    acc00 = wmma_bf16(a0, b0, acc00);
    acc01 = wmma_bf16(a0, b1, acc01);
    acc10 = wmma_bf16(a1, b0, acc10);
    acc11 = wmma_bf16(a1, b1, acc11);
  }
  const int c0 = ct * 32 + (lane & 15);
  const int c1 = c0 + 16;
  qkv_store_tile(acc00, rt * 32,      c0, qkv_b, Qb, Kb, Vt);
  qkv_store_tile(acc01, rt * 32,      c1, qkv_b, Qb, Kb, Vt);
  qkv_store_tile(acc10, rt * 32 + 16, c0, qkv_b, Qb, Kb, Vt);
  qkv_store_tile(acc11, rt * 32 + 16, c1, qkv_b, Qb, Kb, Vt);
}

// ---------------------------------------------------------------------------
// Kernel 4: flash attention. One wave per (n, h, 16-query tile); loop over
// 32-key steps: S = Q K^T (4 WMMAs), online softmax (16-lane butterflies),
// P -> bf16 via wave-local LDS bounce (D-layout -> A-layout), O += P V^T
// (4 WMMAs). Never materializes the 1024x1024 score matrix.
// ---------------------------------------------------------------------------
__global__ void __launch_bounds__(128)
flash_attn_kernel(const __bf16* __restrict__ Qb, const __bf16* __restrict__ Kb,
                  const __bf16* __restrict__ Vt, __bf16* __restrict__ Ob) {
  __shared__ __bf16 lds_p[4][16 * 32];
  const int wave = threadIdx.x >> 5;
  const int lane = threadIdx.x & 31;
  const int wid  = blockIdx.x * 4 + wave;
  const int qt = wid & 63;
  const int h  = (wid >> 6) & 7;
  const int n  = wid >> 9;
  if (n >= kN) return;
  const size_t nh = (size_t)(n * kH + h);

  const __bf16* Qbase = Qb + (nh * kS + (size_t)qt * 16) * kD;
  const v16bf aq0 = load_a_frag(Qbase, kD);        // d = 0..31
  const v16bf aq1 = load_a_frag(Qbase + 32, kD);   // d = 32..63

  v8f o0 = {}, o1 = {}, o2 = {}, o3 = {};
  float m_i[8], l_i[8];
#pragma unroll
  for (int r = 0; r < 8; ++r) { m_i[r] = -3.0e38f; l_i[r] = 0.f; }

  __bf16* myp = lds_p[wave];
  const __bf16* Kbase = Kb + nh * kS * kD;
  const __bf16* Vbase = Vt + nh * kD * kS;

  for (int kt = 0; kt < kS / 32; ++kt) {
    const __bf16* K0 = Kbase + (size_t)(kt * 32) * kD;
    v8f s0 = {}, s1 = {};
    s0 = wmma_bf16(aq0, load_bt_frag(K0, kD), s0);                 // keys 0..15
    s0 = wmma_bf16(aq1, load_bt_frag(K0 + 32, kD), s0);
    s1 = wmma_bf16(aq0, load_bt_frag(K0 + 16 * kD, kD), s1);       // keys 16..31
    s1 = wmma_bf16(aq1, load_bt_frag(K0 + 16 * kD + 32, kD), s1);

    float p0[8], p1[8];
#pragma unroll
    for (int r = 0; r < 8; ++r) {
      const float a0 = s0[r] * kScaleInv;
      const float a1 = s1[r] * kScaleInv;
      const float rm = half_reduce_max(fmaxf(a0, a1));
      const float mn = fmaxf(m_i[r], rm);
      const float alpha = __expf(m_i[r] - mn);
      p0[r] = __expf(a0 - mn);
      p1[r] = __expf(a1 - mn);
      const float rs = half_reduce_add(p0[r] + p1[r]);
      l_i[r] = l_i[r] * alpha + rs;
      m_i[r] = mn;
      o0[r] *= alpha; o1[r] *= alpha; o2[r] *= alpha; o3[r] *= alpha;
    }

    // D-layout (lane=col) -> A-layout (lane=row) redistribution through LDS.
#pragma unroll
    for (int r = 0; r < 8; ++r) {
      const int row = r + ((lane >> 4) << 3);
      const int col = lane & 15;
      myp[row * 32 + col]      = (__bf16)p0[r];
      myp[row * 32 + 16 + col] = (__bf16)p1[r];
    }
    __asm volatile("s_wait_dscnt 0" ::: "memory");   // wave-local LDS fence
    const v16bf ap = load_a_frag(myp, 32);

    o0 = wmma_bf16(ap, load_bt_frag(Vbase + kt * 32, kS), o0);
    o1 = wmma_bf16(ap, load_bt_frag(Vbase + 16 * kS + kt * 32, kS), o1);
    o2 = wmma_bf16(ap, load_bt_frag(Vbase + 32 * kS + kt * 32, kS), o2);
    o3 = wmma_bf16(ap, load_bt_frag(Vbase + 48 * kS + kt * 32, kS), o3);
  }

  // Normalize by row sums, emit bf16 O[(n*s+q)][h*64+d] for the proj GEMM.
#pragma unroll
  for (int r = 0; r < 8; ++r) {
    const float inv = 1.f / l_i[r];
    const int row  = n * kS + qt * 16 + r + ((lane >> 4) << 3);
    const int colb = h * kD + (lane & 15);
    __bf16* orow = Ob + (size_t)row * kNI + colb;
    orow[0]  = (__bf16)(o0[r] * inv);
    orow[16] = (__bf16)(o1[r] * inv);
    orow[32] = (__bf16)(o2[r] * inv);
    orow[48] = (__bf16)(o3[r] * inv);
  }
}

// ---------------------------------------------------------------------------
// Kernel 5: proj GEMM (16384x512 @ 512x512) + bias, store transposed f32
// out[n][c][s]. Same 2x2 register blocking as the QKV GEMM.
// ---------------------------------------------------------------------------
__device__ inline void proj_store_tile(const v8f& acc, int mbase, int c,
                                       const float* __restrict__ proj_b,
                                       float* __restrict__ out) {
  const int lane = threadIdx.x & 31;
  const float bias = proj_b[c];
#pragma unroll
  for (int r = 0; r < 8; ++r) {
    const int mg = mbase + r + ((lane >> 4) << 3);
    const int n  = mg >> 10, srow = mg & (kS - 1);
    out[((size_t)n * kNI + c) * kS + srow] = acc[r] + bias;
  }
}

__global__ void __launch_bounds__(128)
proj_gemm_kernel(const __bf16* __restrict__ Ob, const __bf16* __restrict__ wt,
                 const float* __restrict__ proj_b, float* __restrict__ out) {
  const int wid = blockIdx.x * (blockDim.x >> 5) + (threadIdx.x >> 5);
  const int NTC = kNI / 32;                       // 16 col blocks
  const int rt = wid / NTC, ct = wid % NTC;
  if (rt >= (kN * kS) / 32) return;
  const __bf16* arow0 = Ob + (size_t)(rt * 32) * kNI;
  const __bf16* arow1 = arow0 + (size_t)16 * kNI;
  const __bf16* brow0 = wt + (size_t)(ct * 32) * kNI;
  const __bf16* brow1 = brow0 + (size_t)16 * kNI;
  v8f acc00 = {}, acc01 = {}, acc10 = {}, acc11 = {};
#pragma unroll 2
  for (int k0 = 0; k0 < kNI; k0 += 32) {
    const v16bf a0 = load_a_frag(arow0 + k0, kNI);
    const v16bf a1 = load_a_frag(arow1 + k0, kNI);
    const v16bf b0 = load_bt_frag(brow0 + k0, kNI);
    const v16bf b1 = load_bt_frag(brow1 + k0, kNI);
    acc00 = wmma_bf16(a0, b0, acc00);
    acc01 = wmma_bf16(a0, b1, acc01);
    acc10 = wmma_bf16(a1, b0, acc10);
    acc11 = wmma_bf16(a1, b1, acc11);
  }
  const int lane = threadIdx.x & 31;
  const int c0 = ct * 32 + (lane & 15);
  const int c1 = c0 + 16;
  proj_store_tile(acc00, rt * 32,      c0, proj_b, out);
  proj_store_tile(acc01, rt * 32,      c1, proj_b, out);
  proj_store_tile(acc10, rt * 32 + 16, c0, proj_b, out);
  proj_store_tile(acc11, rt * 32 + 16, c1, proj_b, out);
}

// ---------------------------------------------------------------------------
// Host launcher
// ---------------------------------------------------------------------------
extern "C" void kernel_launch(void* const* d_in, const int* in_sizes, int n_in,
                              void* d_out, int out_size, void* d_ws, size_t ws_size,
                              hipStream_t stream) {
  const float* x      = (const float*)d_in[0];
  const float* ln_g   = (const float*)d_in[1];
  const float* ln_b   = (const float*)d_in[2];
  const float* qkv_w  = (const float*)d_in[3];
  const float* qkv_b  = (const float*)d_in[4];
  const float* proj_w = (const float*)d_in[5];
  const float* proj_b = (const float*)d_in[6];
  float* out = (float*)d_out;
  (void)in_sizes; (void)n_in; (void)out_size; (void)ws_size;

  char* ws = (char*)d_ws;
  size_t off = 0;
  auto alloc = [&](size_t bytes) -> void* {
    void* p = ws + off;
    off += (bytes + 255) & ~(size_t)255;
    return p;
  };
  __bf16* xn      = (__bf16*)alloc((size_t)kN * kS * kNI * 2);       // 16 MB
  __bf16* qkv_wt  = (__bf16*)alloc((size_t)3 * kNI * kNI * 2);       // 1.5 MB
  __bf16* proj_wt = (__bf16*)alloc((size_t)kNI * kNI * 2);           // 0.5 MB
  __bf16* Qb      = (__bf16*)alloc((size_t)kN * kH * kS * kD * 2);   // 16 MB
  __bf16* Kb      = (__bf16*)alloc((size_t)kN * kH * kS * kD * 2);   // 16 MB
  __bf16* Vt      = (__bf16*)alloc((size_t)kN * kH * kD * kS * 2);   // 16 MB
  __bf16* Ob      = (__bf16*)alloc((size_t)kN * kS * kNI * 2);       // 16 MB

  // 1) LN + transpose + bf16
  ln_kernel<<<(kN * kS) / 8, 256, 0, stream>>>(x, ln_g, ln_b, xn);

  // 2) weight convert/transpose
  {
    const int tot = kNI * 3 * kNI + kNI * kNI;
    wconv_kernel<<<(tot + 255) / 256, 256, 0, stream>>>(qkv_w, proj_w, qkv_wt, proj_wt);
  }

  // 3) QKV GEMM -> Q, K, V^T   (32x32 block per wave)
  {
    const int waves = ((kN * kS) / 32) * ((3 * kNI) / 32);   // 24576
    qkv_gemm_kernel<<<waves / 4, 128, 0, stream>>>(xn, qkv_wt, qkv_b, Qb, Kb, Vt);
  }

  // 4) flash attention
  flash_attn_kernel<<<(kN * kH * (kS / 16)) / 4, 128, 0, stream>>>(Qb, Kb, Vt, Ob);

  // 5) proj GEMM + bias + output transpose   (32x32 block per wave)
  {
    const int waves = ((kN * kS) / 32) * (kNI / 32);         // 8192
    proj_gemm_kernel<<<waves / 4, 128, 0, stream>>>(Ob, proj_wt, proj_b, out);
  }
}
